// MaxSigmoidAttnBlock_33749853012629
// MI455X (gfx1250) — compile-verified, matrix operand order checked by reading, not run on hardware
//
#include <hip/hip_runtime.h>
#include <hip/hip_bf16.h>

typedef __attribute__((ext_vector_type(16))) _Float16 v16h;
typedef __attribute__((ext_vector_type(8)))  _Float16 v8h;
typedef __attribute__((ext_vector_type(8)))  float    v8f;

#define CIS 136           // ci stride (halves) in LDS tiles: 272B rows, 16B aligned, bank-spread
#define H_IMG 160
#define W_IMG 160
#define CIN 128
#define CE 64
#define CO 128
#define NG 80
#define CG 512
#define KPROJ 1152        // 9 * 128

// ---- weight repack (f32 -> f16, K-major k=(ky*3+kx)*128+ci) + BN scale/shift precompute ----
__global__ void prep_consts(const float* __restrict__ wproj, const float* __restrict__ wembed,
                            const float* __restrict__ g1, const float* __restrict__ b1,
                            const float* __restrict__ rm1, const float* __restrict__ rv1,
                            const float* __restrict__ g2, const float* __restrict__ b2,
                            const float* __restrict__ rm2, const float* __restrict__ rv2,
                            _Float16* __restrict__ wp16, _Float16* __restrict__ we16,
                            float* __restrict__ sb1, float* __restrict__ tb1,
                            float* __restrict__ sb2, float* __restrict__ tb2) {
  int idx = blockIdx.x * blockDim.x + threadIdx.x;
  if (idx < CO * KPROJ) {
    int co = idx / KPROJ;
    int rem = idx % KPROJ;
    int kk = rem / CIN;           // ky*3+kx
    int ci = rem % CIN;
    int ky = kk / 3, kx = kk % 3;
    wp16[co * KPROJ + kk * CIN + ci] =
        (_Float16)wproj[((co * CIN + ci) * 3 + ky) * 3 + kx];
    return;
  }
  int j = idx - CO * KPROJ;
  if (j < CE * CIN) { we16[j] = (_Float16)wembed[j]; return; }
  j -= CE * CIN;
  if (j < CE) {
    float s = g1[j] * rsqrtf(rv1[j] + 1e-3f);
    sb1[j] = s; tb1[j] = b1[j] - rm1[j] * s;
    return;
  }
  j -= CE;
  if (j < CO) {
    float s = g2[j] * rsqrtf(rv2[j] + 1e-3f);
    sb2[j] = s; tb2[j] = b2[j] - rm2[j] * s;
  }
}

// ---------------- guide_fc: g[b][n][e] = guide[b][n][:] . Wg[e][:] + bg[e] ----------------
__global__ void guide_fc(const float* __restrict__ guide, const float* __restrict__ Wg,
                         const float* __restrict__ bg, float* __restrict__ gbuf) {
  int idx = blockIdx.x * blockDim.x + threadIdx.x;
  if (idx >= 8 * NG * CE) return;
  int e = idx & 63;
  int n = (idx >> 6) % NG;
  int b = idx / (CE * NG);
  const float* gp = guide + ((size_t)(b * NG + n)) * CG;
  const float* wp = Wg + (size_t)e * CG;
  float acc = bg[e];
  #pragma unroll 8
  for (int c = 0; c < CG; ++c) acc += gp[c] * wp[c];
  gbuf[idx] = acc;
}

// ---------------- fused embed 1x1 conv (WMMA) + max-sigmoid attention ----------------
__global__ __launch_bounds__(256) void embed_attn(
    const float* __restrict__ x, const _Float16* __restrict__ we16,
    const float* __restrict__ sb1, const float* __restrict__ tb1,
    const float* __restrict__ gbuf, const float* __restrict__ attn_bias,
    float* __restrict__ awbuf) {
  __shared__ __align__(16) _Float16 xs[32 * CIS];   // [col][ci]
  __shared__ float ef[32 * 68];                     // [px][ce] (pad 68)
  __shared__ float gsh[NG * CE];                    // guide embeddings for this batch
  int w0 = blockIdx.x * 32, h = blockIdx.y, b = blockIdx.z;
  int tid = threadIdx.x;

  // async raw-copy of guide embeddings into LDS (gfx1250 async-to-LDS path, ASYNCcnt)
  for (int i = tid; i < NG * CE; i += 256) {
    unsigned lds_addr = (unsigned)(uintptr_t)(&gsh[i]);
    const float* gp = gbuf + (size_t)b * NG * CE + i;
    asm volatile("global_load_async_to_lds_b32 %0, %1, off"
                 :: "v"(lds_addr), "v"(gp) : "memory");
  }
  for (int i = tid; i < CIN * 32; i += 256) {
    int ci = i >> 5, col = i & 31;
    float v = x[(((size_t)b * CIN + ci) * H_IMG + h) * W_IMG + w0 + col];
    xs[col * CIS + ci] = (_Float16)v;
  }
  asm volatile("s_wait_asynccnt 0x0" ::: "memory");
  __syncthreads();

  int wave = tid >> 5, lane = tid & 31;
  int mt = wave & 3, nt = wave >> 2;       // 4 M-tiles x 2 N-tiles
  int lhalf = lane >> 4, lm = lane & 15;
  int ce0 = mt * 16;
  v8f c = {};
  const _Float16* arow = we16 + (size_t)(ce0 + lm) * CIN;
  const _Float16* brow = xs + (lm + nt * 16) * CIS;
  #pragma unroll
  for (int kk = 0; kk < CIN; kk += 32) {
    int kb = kk + 8 * lhalf;
    v16h a, bf;
    ((v8h*)&a)[0]  = *(const v8h*)(arow + kb);
    ((v8h*)&a)[1]  = *(const v8h*)(arow + kb + 16);
    ((v8h*)&bf)[0] = *(const v8h*)(brow + kb);
    ((v8h*)&bf)[1] = *(const v8h*)(brow + kb + 16);
    c = __builtin_amdgcn_wmma_f32_16x16x32_f16(false, a, false, bf, (short)0, c, false, false);
  }
  int n = lm + nt * 16;
  #pragma unroll
  for (int r = 0; r < 8; ++r) {
    int ce = ce0 + r + 8 * lhalf;
    ef[n * 68 + ce] = c[r] * sb1[ce] + tb1[ce];
  }
  __syncthreads();

  if (tid < 128) {
    int head = tid >> 5, px = tid & 31;
    const float* ep = ef + px * 68 + head * 16;
    float m = -1e30f;
    for (int ng = 0; ng < NG; ++ng) {
      const float* gp = gsh + ng * CE + head * 16;
      float acc = 0.f;
      #pragma unroll
      for (int cc = 0; cc < 16; ++cc) acc += ep[cc] * gp[cc];
      m = fmaxf(m, acc);
    }
    float z = m * 0.25f + attn_bias[head];   // 1/sqrt(16) = 0.25
    awbuf[(((size_t)b * 4 + head) * H_IMG + h) * W_IMG + w0 + px] = 1.0f / (1.0f + __expf(-z));
  }
}

// ---- 3x3 conv (implicit GEMM via WMMA, 2 output rows per block) + BN + per-head gate ----
__global__ __launch_bounds__(256) void conv3_gate(
    const float* __restrict__ x, const _Float16* __restrict__ wp16,
    const float* __restrict__ sb2, const float* __restrict__ tb2,
    const float* __restrict__ awbuf, float* __restrict__ out) {
  __shared__ __align__(16) _Float16 xs[4 * 34 * CIS];   // [ry 0..3][col 0..33][ci], 2-row halo tile
  int w0 = blockIdx.x * 32, h0 = blockIdx.y * 2, b = blockIdx.z;
  int tid = threadIdx.x;

  for (int i = tid; i < CIN * 4 * 34; i += 256) {
    int ci = i / 136;             // 136 = 4*34 positions
    int p = i % 136;
    int ry = p / 34, col = p % 34;
    int hin = h0 + ry - 1, win = w0 + col - 1;
    float v = 0.f;
    if ((unsigned)hin < (unsigned)H_IMG && (unsigned)win < (unsigned)W_IMG)
      v = x[(((size_t)b * CIN + ci) * H_IMG + hin) * W_IMG + win];
    xs[(ry * 34 + col) * CIS + ci] = (_Float16)v;
  }
  __syncthreads();

  int wave = tid >> 5, lane = tid & 31;
  int lhalf = lane >> 4, lm = lane & 15;
  int co0 = wave * 16;                                   // 8 waves cover 128 channels
  v8f c00 = {}, c01 = {}, c10 = {}, c11 = {};            // [row][coltile]
  const _Float16* arow = wp16 + (size_t)(co0 + lm) * KPROJ;

  #pragma unroll 4
  for (int kk = 0; kk < KPROJ; kk += 32) {
    int kyx = kk >> 7;                 // (ky*3+kx)
    int ky = kyx / 3, kx = kyx % 3;
    int kb = (kk & 127) + 8 * lhalf;   // ci offset within this (ky,kx)
    int ab = kk + 8 * lhalf;
    v16h a, b00, b01, b10, b11;
    ((v8h*)&a)[0] = *(const v8h*)(arow + ab);
    ((v8h*)&a)[1] = *(const v8h*)(arow + ab + 16);
    const _Float16* p00 = xs + ((ky * 34) + lm + kx) * CIS + kb;       // row 0
    const _Float16* p10 = p00 + 34 * CIS;                              // row 1 (ky+1)
    ((v8h*)&b00)[0] = *(const v8h*)(p00);
    ((v8h*)&b00)[1] = *(const v8h*)(p00 + 16);
    ((v8h*)&b01)[0] = *(const v8h*)(p00 + 16 * CIS);
    ((v8h*)&b01)[1] = *(const v8h*)(p00 + 16 * CIS + 16);
    ((v8h*)&b10)[0] = *(const v8h*)(p10);
    ((v8h*)&b10)[1] = *(const v8h*)(p10 + 16);
    ((v8h*)&b11)[0] = *(const v8h*)(p10 + 16 * CIS);
    ((v8h*)&b11)[1] = *(const v8h*)(p10 + 16 * CIS + 16);
    c00 = __builtin_amdgcn_wmma_f32_16x16x32_f16(false, a, false, b00, (short)0, c00, false, false);
    c01 = __builtin_amdgcn_wmma_f32_16x16x32_f16(false, a, false, b01, (short)0, c01, false, false);
    c10 = __builtin_amdgcn_wmma_f32_16x16x32_f16(false, a, false, b10, (short)0, c10, false, false);
    c11 = __builtin_amdgcn_wmma_f32_16x16x32_f16(false, a, false, b11, (short)0, c11, false, false);
  }

  int head = co0 >> 5;                                   // one head per wave
  size_t aw0base = (((size_t)b * 4 + head) * H_IMG + h0) * W_IMG + w0;
  float aw00 = awbuf[aw0base + lm];
  float aw01 = awbuf[aw0base + lm + 16];
  float aw10 = awbuf[aw0base + W_IMG + lm];
  float aw11 = awbuf[aw0base + W_IMG + lm + 16];
  #pragma unroll
  for (int r = 0; r < 8; ++r) {
    int co = co0 + r + 8 * lhalf;
    float s = sb2[co];
    float t = tb2[co];
    size_t base = (((size_t)b * CO + co) * H_IMG + h0) * W_IMG + w0;
    out[base + lm]              = (c00[r] * s + t) * aw00;
    out[base + lm + 16]         = (c01[r] * s + t) * aw01;
    out[base + W_IMG + lm]      = (c10[r] * s + t) * aw10;
    out[base + W_IMG + lm + 16] = (c11[r] * s + t) * aw11;
  }
}

extern "C" void kernel_launch(void* const* d_in, const int* in_sizes, int n_in,
                              void* d_out, int out_size, void* d_ws, size_t ws_size,
                              hipStream_t stream) {
  const float* x         = (const float*)d_in[0];
  const float* guide     = (const float*)d_in[1];
  const float* w_embed   = (const float*)d_in[2];
  const float* g1        = (const float*)d_in[3];
  const float* b1        = (const float*)d_in[4];
  const float* rm1       = (const float*)d_in[5];
  const float* rv1       = (const float*)d_in[6];
  const float* Wg        = (const float*)d_in[7];
  const float* bg        = (const float*)d_in[8];
  const float* attn_bias = (const float*)d_in[9];
  const float* w_proj    = (const float*)d_in[10];
  const float* g2        = (const float*)d_in[11];
  const float* b2        = (const float*)d_in[12];
  const float* rm2       = (const float*)d_in[13];
  const float* rv2       = (const float*)d_in[14];
  float* out = (float*)d_out;

  char* ws = (char*)d_ws;
  _Float16* wp16 = (_Float16*)(ws + 0);        // 128*1152*2 = 294912 B
  _Float16* we16 = (_Float16*)(ws + 294912);   // 64*128*2   = 16384 B
  float*    gbuf = (float*)(ws + 311296);      // 8*80*64*4  = 163840 B
  float*    awbuf= (float*)(ws + 475136);      // 8*4*160*160*4 = 3276800 B
  float*    sb1  = (float*)(ws + 3751936);     // 64*4
  float*    tb1  = (float*)(ws + 3752192);     // 64*4
  float*    sb2  = (float*)(ws + 3752448);     // 128*4
  float*    tb2  = (float*)(ws + 3752960);     // 128*4

  int prepN = CO * KPROJ + CE * CIN + CE + CO;
  prep_consts<<<(prepN + 255) / 256, 256, 0, stream>>>(
      w_proj, w_embed, g1, b1, rm1, rv1, g2, b2, rm2, rv2,
      wp16, we16, sb1, tb1, sb2, tb2);
  guide_fc<<<(8 * NG * CE + 255) / 256, 256, 0, stream>>>(guide, Wg, bg, gbuf);

  dim3 grid_e(W_IMG / 32, H_IMG, 8);
  embed_attn<<<grid_e, 256, 0, stream>>>(x, we16, sb1, tb1, gbuf, attn_bias, awbuf);

  dim3 grid_c(W_IMG / 32, H_IMG / 2, 8);
  conv3_gate<<<grid_c, 256, 0, stream>>>(x, wp16, sb2, tb2, awbuf, out);
}
